// HoleNN_12232066859166
// MI455X (gfx1250) — matrix operand order repaired and checked by reading, not compiled, not run on hardware
//
#include <hip/hip_runtime.h>

typedef __attribute__((ext_vector_type(16))) _Float16 v16h;
typedef __attribute__((ext_vector_type(8)))  _Float16 v8h;
typedef __attribute__((ext_vector_type(4)))  _Float16 v4h;
typedef __attribute__((ext_vector_type(8)))  float    v8f;
typedef __attribute__((ext_vector_type(4)))  unsigned int u32x4;
typedef __attribute__((ext_vector_type(8)))  int      i32x8;
typedef __attribute__((ext_vector_type(4)))  int      i32x4;

#define FEAT     512
#define MT       64            // batch rows per workgroup
#define ASTR     520           // A/C staging stride in halfs (1040B, 16B-aligned rows)
#define HSTR     513           // HA/HB stride (scalar access only; odd => bank spread)
#define WSTR     136           // W slab stride in halfs (272B, 16B-aligned, 4-bank shift)
#define KSLAB    128           // K depth staged per W slab
#define NSLABS   (FEAT / KSLAB)
#define NCHUNK   128           // output columns per chunk (4 wave col tiles x 32 cols)
#define NCHUNKS  (FEAT / NCHUNK)
#define WSLAB_HALFS (NCHUNK * WSTR)                 // 17408

// LDS layout (offsets in halfs)
#define OFF_A    0
#define OFF_W0   (MT * ASTR)                        // 33280
#define OFF_W1   (OFF_W0 + WSLAB_HALFS)             // 50688
#define OFF_HA   (OFF_W1 + WSLAB_HALFS)             // 68096
#define OFF_HB   (OFF_HA + MT * HSTR)               // 100928
#define SMEM_HALFS (OFF_HB + MT * HSTR)             // 133760
#define SMEM_BYTES (SMEM_HALFS * 2)                 // 267520 B <= 320KB WGP LDS

// combined scale: 0.5 (Hartley combine) * 1/N (inverse DHT) * 1/sqrt(N)
#define COMB_SCALE 4.3158373e-05f

#if defined(__has_builtin)
#if __has_builtin(__builtin_amdgcn_tensor_load_to_lds) && __has_builtin(__builtin_amdgcn_s_wait_tensorcnt)
#define HAVE_TDM 1
#endif
#endif

// ---------------------------------------------------------------------------
// W[j,k] = cas(2*pi*j*k/512) = cos + sin  (symmetric, shared by all rows)
// ---------------------------------------------------------------------------
__global__ void cas_init_kernel(_Float16* __restrict__ W) {
    int idx = blockIdx.x * blockDim.x + threadIdx.x;   // 0 .. 512*512-1
    int j = idx >> 9;
    int k = idx & (FEAT - 1);
    int p = (j * k) & (FEAT - 1);                      // exact mod-512 phase
    float th = (float)p * 1.2271846e-02f;              // 2*pi/512
    W[idx] = (_Float16)(cosf(th) + sinf(th));
}

// ---------------------------------------------------------------------------
// Fragment loaders (ISA 7.12.2 16-bit layouts, wave32)
// ---------------------------------------------------------------------------
__device__ __forceinline__ v16h load_fragA(const _Float16* sm, int rowBase, int k0, int lane) {
    int row  = rowBase + (lane & 15);
    int koff = (lane >> 4) << 3;                       // 0 or 8
    const _Float16* p = sm + row * ASTR + k0 + koff;
    v8h lo = *(const v8h*)(p);
    v8h hi = *(const v8h*)(p + 16);
    return __builtin_shufflevector(lo, hi, 0,1,2,3,4,5,6,7,8,9,10,11,12,13,14,15);
}

__device__ __forceinline__ v16h load_fragB(const _Float16* smW, int colBase, int kl, int lane) {
    int col  = colBase + (lane & 15);
    int koff = (lane >> 4) << 3;                       // 0 or 8
    const _Float16* p = smW + col * WSTR + kl + koff;
    v8h lo = *(const v8h*)(p);
    v8h hi = *(const v8h*)(p + 16);
    return __builtin_shufflevector(lo, hi, 0,1,2,3,4,5,6,7,8,9,10,11,12,13,14,15);
}

#define WMMA_F16(A, B, C) \
    __builtin_amdgcn_wmma_f32_16x16x32_f16(false, (A), false, (B), (short)0, (C), false, false)

#ifdef HAVE_TDM
// ---------------------------------------------------------------------------
// TDM: DMA one W slab (128 rows x 128 halfs) into LDS with pad to WSTR=136.
// D# group0: count=1 | lds_addr | global_addr | type=2
// D# group1: data_size=2B, pad_enable, pad_interval=5 (256B), pad_amount=3 (16B),
//            tensor 512x512 stride 512, tile 128x128
// ---------------------------------------------------------------------------
__device__ __forceinline__ void tdm_load_w(const _Float16* Wg, int chunkBase, int ks,
                                           unsigned ldsByteOff) {
    unsigned long long ga =
        (unsigned long long)(uintptr_t)(Wg + chunkBase * FEAT + ks);
    u32x4 g0;
    g0[0] = 1u;                                       // count=1 (valid user D#)
    g0[1] = ldsByteOff;                               // lds_addr (bytes)
    g0[2] = (unsigned)ga;                             // global_addr[31:0]
    g0[3] = (unsigned)(ga >> 32) | (2u << 30);        // global_addr[56:32] | type=2
    i32x8 g1;
    g1[0] = (int)((1u << 16) | (1u << 20) | (5u << 22) | (3u << 25));
    g1[1] = (int)(512u << 16);                        // tensor_dim0 = 512
    g1[2] = (int)(512u << 16);                        // tensor_dim1 = 512
    g1[3] = (int)(128u << 16);                        // tile_dim0  = 128
    g1[4] = 128;                                      // tile_dim1  = 128
    g1[5] = 512;                                      // tensor_dim0_stride = 512
    g1[6] = 0;
    g1[7] = 0;
    i32x4 z4 = {0, 0, 0, 0};
#if __clang_major__ >= 23
    i32x8 z8 = {0, 0, 0, 0, 0, 0, 0, 0};
    __builtin_amdgcn_tensor_load_to_lds(g0, g1, z4, z4, z8, 0);
#else
    __builtin_amdgcn_tensor_load_to_lds(g0, g1, z4, z4, 0);
#endif
}
#endif

// ---------------------------------------------------------------------------
// Fused: HA = A*W ; HB = B*W ; Hartley combine ; out = C*W (scale folded into C)
// ---------------------------------------------------------------------------
__global__ void __launch_bounds__(256, 1)
hartley_corr_kernel(const float* __restrict__ A, const float* __restrict__ B,
                    const _Float16* __restrict__ W, float* __restrict__ out) {
    extern __shared__ _Float16 smem[];
    _Float16* smA  = smem + OFF_A;    // input staging (A, then B, then combined C)
    _Float16* smW0 = smem + OFF_W0;   // W slab ping
    _Float16* smW1 = smem + OFF_W1;   // W slab pong
    _Float16* smHA = smem + OFF_HA;
    _Float16* smHB = smem + OFF_HB;

    const int  tid  = threadIdx.x;
    const int  lane = tid & 31;
    const int  wv   = tid >> 5;       // wave32: 8 waves
    const int  rt   = wv & 1;         // macro row tile: rows rt*32 .. rt*32+31
    const int  ct   = wv >> 1;        // macro col tile: cols ct*32 .. ct*32+31
    const long rowBlock = (long)blockIdx.x * MT;

    auto load_input = [&](const float* g) {
        const float* base = g + rowBlock * FEAT;      // 64 contiguous rows
        for (int i = tid; i < MT * FEAT / 4; i += 256) {
            int flat = i << 2;
            float4 v = *(const float4*)(base + flat);
            int r = flat >> 9, k = flat & (FEAT - 1);
            v4h h;
            h[0] = (_Float16)v.x; h[1] = (_Float16)v.y;
            h[2] = (_Float16)v.z; h[3] = (_Float16)v.w;
            *(v4h*)(smA + r * ASTR + k) = h;
        }
    };

#ifndef HAVE_TDM
    auto load_w_slab = [&](int chunkBase, int ks, _Float16* dst) {
        for (int i = tid; i < NCHUNK * KSLAB / 8; i += 256) {
            int flat = i << 3;
            int r = flat >> 7, k = flat & (KSLAB - 1);
            v8h w = *(const v8h*)(W + (chunkBase + r) * FEAT + ks + k);
            *(v8h*)(dst + r * WSTR + k) = w;
        }
        if (ks + KSLAB < FEAT)
            __builtin_prefetch(W + chunkBase * FEAT + ks + KSLAB, 0, 0);
    };
#endif

    auto compute_slab = [&](const _Float16* wbuf, int ks,
                            v8f& acc00, v8f& acc01, v8f& acc10, v8f& acc11) {
#pragma unroll
        for (int kl = 0; kl < KSLAB; kl += 32) {
            v16h a0 = load_fragA(smA, rt * 32,      ks + kl, lane);
            v16h a1 = load_fragA(smA, rt * 32 + 16, ks + kl, lane);
            v16h b0 = load_fragB(wbuf, ct * 32,      kl, lane);
            v16h b1 = load_fragB(wbuf, ct * 32 + 16, kl, lane);
            acc00 = WMMA_F16(a0, b0, acc00);
            acc01 = WMMA_F16(a0, b1, acc01);
            acc10 = WMMA_F16(a1, b0, acc10);
            acc11 = WMMA_F16(a1, b1, acc11);
        }
    };

    // one GEMM pass: smA[64x512] x W[512x512] -> HA/HB (fp16 LDS) or out (f32 global)
    auto gemm_pass = [&](_Float16* smOut, bool toGlobal) {
        for (int c = 0; c < NCHUNKS; ++c) {
            const int chunkBase = c * NCHUNK;
            v8f acc00 = {}, acc01 = {}, acc10 = {}, acc11 = {};
#ifdef HAVE_TDM
            __syncthreads();                          // all buffers free
            if (wv == 0) tdm_load_w(W, chunkBase, 0, OFF_W0 * 2u);
            for (int s = 0; s < NSLABS; ++s) {
                if (wv == 0) __builtin_amdgcn_s_wait_tensorcnt(0);  // slab s landed
                __syncthreads();                      // slab s visible; s-1 compute done
                if (wv == 0 && s + 1 < NSLABS)        // overlap DMA of s+1 with compute s
                    tdm_load_w(W, chunkBase, (s + 1) * KSLAB,
                               (((s + 1) & 1) ? OFF_W1 : OFF_W0) * 2u);
                compute_slab((s & 1) ? smW1 : smW0, s * KSLAB, acc00, acc01, acc10, acc11);
            }
#else
            for (int s = 0; s < NSLABS; ++s) {
                __syncthreads();
                load_w_slab(chunkBase, s * KSLAB, smW0);
                __syncthreads();
                compute_slab(smW0, s * KSLAB, acc00, acc01, acc10, acc11);
            }
#endif
            // C/D layout: lane<16 -> N=lane, M=vgpr; lane>=16 -> N=lane-16, M=8+vgpr
            const int n  = lane & 15;
            const int mh = (lane >> 4) * 8;
            if (toGlobal) {
                long row0 = rowBlock + rt * 32 + mh;
                float* p = out + row0 * FEAT + chunkBase + ct * 32 + n;
#pragma unroll
                for (int v = 0; v < 8; ++v) {
                    p[v * FEAT]             = acc00[v];
                    p[v * FEAT + 16]        = acc01[v];
                    p[(v + 16) * FEAT]      = acc10[v];
                    p[(v + 16) * FEAT + 16] = acc11[v];
                }
            } else {
                _Float16* p = smOut + (rt * 32 + mh) * HSTR + chunkBase + ct * 32 + n;
#pragma unroll
                for (int v = 0; v < 8; ++v) {
                    p[v * HSTR]             = (_Float16)acc00[v];
                    p[v * HSTR + 16]        = (_Float16)acc01[v];
                    p[(v + 16) * HSTR]      = (_Float16)acc10[v];
                    p[(v + 16) * HSTR + 16] = (_Float16)acc11[v];
                }
            }
        }
    };

    // ---- stage 1: HA = A * W
    load_input(A);
    gemm_pass(smHA, false);
    __syncthreads();

    // ---- stage 2: HB = B * W
    load_input(B);
    gemm_pass(smHB, false);
    __syncthreads();

    // ---- stage 3: Hartley-domain correlation combine -> smA (fp16)
    for (int i = tid; i < MT * FEAT; i += 256) {
        int r = i >> 9, k = i & (FEAT - 1);
        int rk = (FEAT - k) & (FEAT - 1);
        float hak  = (float)smHA[r * HSTR + k];
        float hark = (float)smHA[r * HSTR + rk];
        float hbk  = (float)smHB[r * HSTR + k];
        float hbrk = (float)smHB[r * HSTR + rk];
        float cv = (hark * (hbk + hbrk) + hak * (hbk - hbrk)) * COMB_SCALE;
        smA[r * ASTR + k] = (_Float16)cv;
    }
    __syncthreads();

    // ---- stage 4: out = C * W
    gemm_pass(nullptr, true);
}

// ---------------------------------------------------------------------------
extern "C" void kernel_launch(void* const* d_in, const int* in_sizes, int n_in,
                              void* d_out, int out_size, void* d_ws, size_t ws_size,
                              hipStream_t stream) {
    const float* A = (const float*)d_in[0];
    const float* B = (const float*)d_in[1];
    float* outp = (float*)d_out;
    _Float16* W = (_Float16*)d_ws;                    // 512KB fp16 DHT matrix
    int batch = in_sizes[0] / FEAT;

    (void)hipFuncSetAttribute(reinterpret_cast<const void*>(&hartley_corr_kernel),
                              hipFuncAttributeMaxDynamicSharedMemorySize, SMEM_BYTES);

    cas_init_kernel<<<(FEAT * FEAT) / 256, 256, 0, stream>>>(W);
    hartley_corr_kernel<<<batch / MT, 256, SMEM_BYTES, stream>>>(A, B, W, outp);
}
